// Linear_6605659701450
// MI455X (gfx1250) — 8-bit matrix operand order repaired and checked by reading, compiled, not run on hardware
//
#include <hip/hip_runtime.h>
#include <math.h>

#define TOK   8192
#define KDIM  2048
#define NDIM  8192
#define LDSS  80

typedef __attribute__((ext_vector_type(8))) int v8i;

__device__ __forceinline__ void async_ld16(void* ldsDst, const void* gSrc) {
    unsigned ldsOff = (unsigned)(unsigned long long)(uintptr_t)ldsDst;
    asm volatile("global_load_async_to_lds_b128 %0, %1, off"
                 :: "v"(ldsOff), "v"(gSrc)
                 : "memory");
}
__device__ __forceinline__ void async_wait0() {
    asm volatile("s_wait_asynccnt 0x0" ::: "memory");
}

__global__ void zero_ws_kernel(float* p) { p[0] = 0.0f; }

__global__ void wabs_sum_kernel(const float4* __restrict__ w,
                                float* __restrict__ sum, int n4) {
    __shared__ float red[256];
    float s = 0.0f;
    for (int i = blockIdx.x * blockDim.x + threadIdx.x; i < n4;
         i += gridDim.x * blockDim.x) {
        float4 v = w[i];
        s += fabsf(v.x) + fabsf(v.y) + fabsf(v.z) + fabsf(v.w);
    }
    red[threadIdx.x] = s;
    __syncthreads();
    for (int o = 128; o > 0; o >>= 1) {
        if (threadIdx.x < o) red[threadIdx.x] += red[threadIdx.x + o];
        __syncthreads();
    }
    if (threadIdx.x == 0) atomicAdd(sum, red[0]);
}

__global__ void wquant_kernel(const float4* __restrict__ w,
                              char* __restrict__ wq,
                              const float* __restrict__ sum, int n4) {
    int i = blockIdx.x * blockDim.x + threadIdx.x;
    if (i >= n4) return;
    float mean  = sum[0] * (1.0f / (float)((size_t)NDIM * KDIM));
    float scale = 1.0f / fmaxf(mean, 1e-5f);
    float4 v = w[i];
    int q0 = (int)rintf(v.x * scale); q0 = min(max(q0, -1), 1);
    int q1 = (int)rintf(v.y * scale); q1 = min(max(q1, -1), 1);
    int q2 = (int)rintf(v.z * scale); q2 = min(max(q2, -1), 1);
    int q3 = (int)rintf(v.w * scale); q3 = min(max(q3, -1), 1);
    ((int*)wq)[i] = (q0 & 0xff) | ((q1 & 0xff) << 8) |
                    ((q2 & 0xff) << 16) | ((q3 & 0xff) << 24);
}

__global__ void actquant_kernel(const float* __restrict__ x,
                                char* __restrict__ xq,
                                float* __restrict__ dq_tok) {
    __shared__ float rs[256];
    __shared__ float rm[256];
    const int tok = blockIdx.x;
    const int t   = threadIdx.x;
    const float4* row = (const float4*)(x + (size_t)tok * KDIM);

    float4 v0 = row[t];
    float4 v1 = row[t + 256];

    float ss = v0.x*v0.x + v0.y*v0.y + v0.z*v0.z + v0.w*v0.w
             + v1.x*v1.x + v1.y*v1.y + v1.z*v1.z + v1.w*v1.w;
    float mx = fmaxf(fmaxf(fmaxf(fabsf(v0.x), fabsf(v0.y)),
                           fmaxf(fabsf(v0.z), fabsf(v0.w))),
                     fmaxf(fmaxf(fabsf(v1.x), fabsf(v1.y)),
                           fmaxf(fabsf(v1.z), fabsf(v1.w))));
    rs[t] = ss; rm[t] = mx;
    __syncthreads();
    for (int o = 128; o > 0; o >>= 1) {
        if (t < o) { rs[t] += rs[t + o]; rm[t] = fmaxf(rm[t], rm[t + o]); }
        __syncthreads();
    }
    const float eps   = 1.1920928955078125e-7f;
    float rnorm = rsqrtf(rs[0] * (1.0f / (float)KDIM) + eps);
    float maxn  = rm[0] * rnorm;
    float dq    = fmaxf(maxn, 1e-5f) * (1.0f / 127.0f);
    float sc    = rnorm / dq;

    int q; int pk0, pk1;
    q = min(max((int)rintf(v0.x * sc), -128), 127); pk0 =  (q & 0xff);
    q = min(max((int)rintf(v0.y * sc), -128), 127); pk0 |= (q & 0xff) << 8;
    q = min(max((int)rintf(v0.z * sc), -128), 127); pk0 |= (q & 0xff) << 16;
    q = min(max((int)rintf(v0.w * sc), -128), 127); pk0 |= (q & 0xff) << 24;
    q = min(max((int)rintf(v1.x * sc), -128), 127); pk1 =  (q & 0xff);
    q = min(max((int)rintf(v1.y * sc), -128), 127); pk1 |= (q & 0xff) << 8;
    q = min(max((int)rintf(v1.z * sc), -128), 127); pk1 |= (q & 0xff) << 16;
    q = min(max((int)rintf(v1.w * sc), -128), 127); pk1 |= (q & 0xff) << 24;

    int* dst = (int*)(xq + (size_t)tok * KDIM);
    dst[t]       = pk0;
    dst[t + 256] = pk1;
    if (t == 0) dq_tok[tok] = dq;
}

__global__ __launch_bounds__(256) void gemm_i8_kernel(
        const char* __restrict__ xq, const char* __restrict__ wq,
        const float* __restrict__ dq_tok, const float* __restrict__ sum,
        float* __restrict__ out) {
    __shared__ unsigned char sA[2][128 * LDSS];
    __shared__ unsigned char sB[2][256 * LDSS];

    const int t     = threadIdx.x;
    const int lane  = t & 31;
    const int wave  = t >> 5;
    const int waveM = wave & 1;
    const int waveN = wave >> 1;
    const int half  = lane >> 4;
    const int l16   = lane & 15;

    const int tokBase = blockIdx.y * 128;
    const int nBase   = blockIdx.x * 256;

    const int lrow = t >> 2;
    const int lcol = (t & 3) * 16;

    const v8i vzero = {0, 0, 0, 0, 0, 0, 0, 0};
    v8i acc[4][4];
#pragma unroll
    for (int mi = 0; mi < 4; ++mi)
#pragma unroll
        for (int ni = 0; ni < 4; ++ni) acc[mi][ni] = vzero;

    const char* aPtr = xq + (size_t)tokBase * KDIM;
    const char* bPtr = wq + (size_t)nBase * KDIM;

#pragma unroll
    for (int j = 0; j < 2; ++j)
        async_ld16(&sA[0][(lrow + 64 * j) * LDSS + lcol],
                   aPtr + (size_t)(lrow + 64 * j) * KDIM + lcol);
#pragma unroll
    for (int j = 0; j < 4; ++j)
        async_ld16(&sB[0][(lrow + 64 * j) * LDSS + lcol],
                   bPtr + (size_t)(lrow + 64 * j) * KDIM + lcol);

    const int KSTEPS = KDIM / 64;
    for (int ks = 0; ks < KSTEPS; ++ks) {
        async_wait0();
        __syncthreads();
        const int buf = ks & 1;

        if (ks + 1 < KSTEPS) {
            const int kb = (ks + 1) * 64;
#pragma unroll
            for (int j = 0; j < 2; ++j)
                async_ld16(&sA[buf ^ 1][(lrow + 64 * j) * LDSS + lcol],
                           aPtr + (size_t)(lrow + 64 * j) * KDIM + kb + lcol);
#pragma unroll
            for (int j = 0; j < 4; ++j)
                async_ld16(&sB[buf ^ 1][(lrow + 64 * j) * LDSS + lcol],
                           bPtr + (size_t)(lrow + 64 * j) * KDIM + kb + lcol);
        }

        v8i af[4];
#pragma unroll
        for (int mi = 0; mi < 4; ++mi) {
            const int arow = waveM * 64 + mi * 16 + l16;
#pragma unroll
            for (int p = 0; p < 4; ++p) {
                int2 d = *(const int2*)&sA[buf][arow * LDSS + (p >> 1) * 32 + half * 16 + (p & 1) * 8];
                af[mi][2 * p]     = d.x;
                af[mi][2 * p + 1] = d.y;
            }
        }

#pragma unroll
        for (int ni = 0; ni < 4; ++ni) {
            const int brow = waveN * 64 + ni * 16 + l16;
            int4 d0 = *(const int4*)&sB[buf][brow * LDSS + half * 16];
            int4 d1 = *(const int4*)&sB[buf][brow * LDSS + 32 + half * 16];
            v8i bf;
            bf[0] = d0.x; bf[1] = d0.y; bf[2] = d0.z; bf[3] = d0.w;
            bf[4] = d1.x; bf[5] = d1.y; bf[6] = d1.z; bf[7] = d1.w;
#pragma unroll
            for (int mi = 0; mi < 4; ++mi)
                acc[mi][ni] = __builtin_amdgcn_wmma_i32_16x16x64_iu8(
                     true, af[mi],  true, bf,
                    acc[mi][ni],  false,  false);
        }
    }

    const float dqw = fmaxf(sum[0] * (1.0f / (float)((size_t)NDIM * KDIM)), 1e-5f);
#pragma unroll
    for (int mi = 0; mi < 4; ++mi) {
#pragma unroll
        for (int ni = 0; ni < 4; ++ni) {
            const int n = nBase + waveN * 64 + ni * 16 + l16;
#pragma unroll
            for (int r = 0; r < 8; ++r) {
                const int tokRow = tokBase + waveM * 64 + mi * 16 + half * 8 + r;
                out[(size_t)tokRow * NDIM + n] =
                    (float)acc[mi][ni][r] * dq_tok[tokRow] * dqw;
            }
        }
    }
}

extern "C" void kernel_launch(void* const* d_in, const int* in_sizes, int n_in,
                              void* d_out, int out_size, void* d_ws, size_t ws_size,
                              hipStream_t stream) {
    const float* x = (const float*)d_in[0];
    const float* w = (const float*)d_in[1];
    float* out = (float*)d_out;

    char*  ws     = (char*)d_ws;
    float* sum    = (float*)ws;
    float* dq_tok = (float*)(ws + 256);
    char*  xq     = ws + 33280;
    char*  wq     = xq + (size_t)TOK * KDIM;

    const int n4 = (NDIM * KDIM) / 4;

    zero_ws_kernel<<<1, 1, 0, stream>>>(sum);
    wabs_sum_kernel<<<1024, 256, 0, stream>>>((const float4*)w, sum, n4);
    wquant_kernel<<<(n4 + 255) / 256, 256, 0, stream>>>((const float4*)w, wq, sum, n4);
    actquant_kernel<<<TOK, 256, 0, stream>>>(x, xq, dq_tok);

    dim3 grid(NDIM / 256, TOK / 128);
    gemm_i8_kernel<<<grid, 256, 0, stream>>>(xq, wq, dq_tok, sum, out);
}
